// KnowledgeConsolidationBlock_90400471646459
// MI455X (gfx1250) — matrix operand order repaired and checked by reading, not compiled, not run on hardware
//
#include <hip/hip_runtime.h>
#include <hip/hip_bf16.h>
#include <math.h>

typedef __attribute__((ext_vector_type(16))) _Float16 v16h;
typedef __attribute__((ext_vector_type(8)))  _Float16 v8h;
typedef __attribute__((ext_vector_type(4)))  _Float16 v4h;
typedef __attribute__((ext_vector_type(8)))  float    v8f;
typedef __attribute__((ext_vector_type(4)))  float    v4f;

#define WMMA_F32_F16(a, b, c) \
    __builtin_amdgcn_wmma_f32_16x16x32_f16(false, (a), false, (b), (short)0, (c), false, false)

// --- CDNA5 16-bit fragment layout inverses (element -> (lane, slot)) -------
// A-matrix 16x32 (m rows, k cols): lane = 16*((k&15)>>3) + m
__device__ __forceinline__ int laneA(int m, int k) { return (((k & 15) >> 3) << 4) + m; }
__device__ __forceinline__ int slotA(int k)        { return (k & 7) | ((k >> 4) << 3); }
// B-matrix 32x16 (k rows, n cols): lane = 16*((k&15)>>3) + n
__device__ __forceinline__ int laneB(int k, int n) { return (((k & 15) >> 3) << 4) + n; }
__device__ __forceinline__ int slotB(int k)        { return (k & 7) | ((k >> 4) << 3); }

// Load one lane's 16-half fragment (two ds_load_b128) from a 24-half row.
__device__ __forceinline__ v16h load_frag(const _Float16* row) {
    v8h lo = *(const v8h*)(row);
    v8h hi = *(const v8h*)(row + 8);
    return __builtin_shufflevector(lo, hi, 0, 1, 2, 3, 4, 5, 6, 7,
                                          8, 9, 10, 11, 12, 13, 14, 15);
}

__device__ __forceinline__ v4h cvt4(float a, float b, float c, float d) {
    v4h r;
    r[0] = (_Float16)a; r[1] = (_Float16)b; r[2] = (_Float16)c; r[3] = (_Float16)d;
    return r;
}

// ---------------------------------------------------------------------------
// GEMM: out[b] = W(MxK) @ X[b](KxN), fp32 in/out, f16 WMMA compute.
// MODE 0: plain   MODE 1: +bias, exact GELU   MODE 2: +bias
// Block 256 threads (8 waves). Tile 64(M) x 128(N). K stepped by 32 with
// double-buffered LDS: next chunk's global loads overlap current WMMAs.
// ---------------------------------------------------------------------------
template <int MODE>
__global__ __launch_bounds__(256)
void gemm_wmma_kernel(const float* __restrict__ W, const float* __restrict__ X,
                      const float* __restrict__ bias, float* __restrict__ out,
                      int M, int N, int K) {
    __shared__ __align__(16) _Float16 sWA[2][4][32][24];  // fragment-ordered A
    __shared__ __align__(16) _Float16 sXB[2][8][32][24];  // fragment-ordered B

    const int tid  = threadIdx.x;
    const int wave = tid >> 5;
    const int lane = tid & 31;
    const int g    = lane >> 4;
    const int ln   = lane & 15;
    const int wm   = wave & 1;
    const int wn   = wave >> 1;
    const int m0   = blockIdx.y * 64;
    const int n0   = blockIdx.x * 128;
    const size_t bX = (size_t)blockIdx.z * K * N;

    v8f acc[2][2] = {{{}, {}}, {{}, {}}};
    v4f rw[2];           // prefetched W quads
    float rx[4][4];      // prefetched X quads

    auto load_regs = [&](int k0) {
        #pragma unroll
        for (int i = 0; i < 2; ++i) {
            int e  = tid + i * 256;            // 512 quads (64x32 W chunk)
            int mi = e >> 3, kq = (e & 7) * 4;
            rw[i] = *(const v4f*)&W[(size_t)(m0 + mi) * K + k0 + kq];
        }
        #pragma unroll
        for (int i = 0; i < 4; ++i) {
            int e = tid + i * 256;             // 1024 quads (32x128 X chunk)
            int n = e & 127, kq = (e >> 7) * 4;
            const float* xp = &X[bX + (size_t)(k0 + kq) * N + n0 + n];
            rx[i][0] = xp[0];
            rx[i][1] = xp[(size_t)N];
            rx[i][2] = xp[(size_t)2 * N];
            rx[i][3] = xp[(size_t)3 * N];
        }
    };
    auto store_regs = [&](int buf) {
        #pragma unroll
        for (int i = 0; i < 2; ++i) {
            int e  = tid + i * 256;
            int mi = e >> 3, kq = (e & 7) * 4;
            *(v4h*)&sWA[buf][mi >> 4][laneA(mi & 15, kq)][slotA(kq)] =
                cvt4(rw[i][0], rw[i][1], rw[i][2], rw[i][3]);
        }
        #pragma unroll
        for (int i = 0; i < 4; ++i) {
            int e = tid + i * 256;
            int n = e & 127, kq = (e >> 7) * 4;
            *(v4h*)&sXB[buf][n >> 4][laneB(kq, n & 15)][slotB(kq)] =
                cvt4(rx[i][0], rx[i][1], rx[i][2], rx[i][3]);
        }
    };

    load_regs(0);
    store_regs(0);

    for (int k0 = 0; k0 < K; k0 += 32) {
        const int cur = (k0 >> 5) & 1;
        __syncthreads();                       // publishes buf 'cur', retires 'cur^1'
        const bool has_next = (k0 + 32) < K;
        if (has_next) load_regs(k0 + 32);      // VMEM in flight during compute

        v16h At[2], Bt[2];
        #pragma unroll
        for (int ms = 0; ms < 2; ++ms) At[ms] = load_frag(&sWA[cur][2 * wm + ms][lane][0]);
        #pragma unroll
        for (int ns = 0; ns < 2; ++ns) Bt[ns] = load_frag(&sXB[cur][2 * wn + ns][lane][0]);
        #pragma unroll
        for (int ms = 0; ms < 2; ++ms)
            #pragma unroll
            for (int ns = 0; ns < 2; ++ns)
                acc[ms][ns] = WMMA_F32_F16(At[ms], Bt[ns], acc[ms][ns]);

        if (has_next) store_regs(cur ^ 1);     // safe: cur^1 readers done at barrier
    }

    #pragma unroll
    for (int ms = 0; ms < 2; ++ms)
        #pragma unroll
        for (int ns = 0; ns < 2; ++ns)
            #pragma unroll
            for (int v = 0; v < 8; ++v) {
                int m = m0 + (2 * wm + ms) * 16 + v + 8 * g;
                int n = n0 + (2 * wn + ns) * 16 + ln;
                float val = acc[ms][ns][v];
                if (MODE >= 1) val += bias[m];
                if (MODE == 1) val = 0.5f * val * (1.0f + erff(val * 0.70710678118654752f));
                out[((size_t)blockIdx.z * M + m) * N + n] = val;
            }
}

// ---------------------------------------------------------------------------
// Flash attention. qkv layout (B, 1152, 1024): q rows [0,384), k [384,768),
// v [768,1152); within each region, row = head*48 + d. d=48 padded to 64.
// Grid: (N/64, heads, B). Block 128 threads = 4 waves, 16 queries per wave.
// Key loop double-buffered like the GEMM.
// ---------------------------------------------------------------------------
__global__ __launch_bounds__(128)
void attn_kernel(const float* __restrict__ qkv, float* __restrict__ out, float scale) {
    __shared__ __align__(16) _Float16 sQA[4][2][32][24];    // [qtile][kchunk] A of Q
    __shared__ __align__(16) _Float16 sKB[2][2][2][32][24]; // [buf][kchunk][ntile] B of K
    __shared__ __align__(16) _Float16 sVB[2][3][32][24];    // [buf][dtile] B of V^T
    __shared__ __align__(16) _Float16 sPA[4][32][24];       // per-wave A of P

    const int tid  = threadIdx.x;
    const int w    = tid >> 5;
    const int lane = tid & 31;
    const int g    = lane >> 4;
    const int ln   = lane & 15;
    const int b    = blockIdx.z;
    const int head = blockIdx.y;
    const int qb   = blockIdx.x * 64;

    const float* qp = qkv + ((size_t)b * 1152 + head * 48) * 1024;
    const float* kp = qp + (size_t)384 * 1024;
    const float* vp = qp + (size_t)768 * 1024;

    // Stage Q (64 queries x 64 d, zero-padded d>=48), fragment-ordered
    #pragma unroll
    for (int i = 0; i < 8; ++i) {
        int e  = tid + i * 128;
        int nl = e & 63, dq = (e >> 6) * 4;
        v4h p;
        if (dq < 48) {
            const float* q4 = &qp[(size_t)dq * 1024 + qb + nl];
            p = cvt4(q4[0], q4[1024], q4[2048], q4[3072]);
        } else {
            p = cvt4(0.f, 0.f, 0.f, 0.f);
        }
        int kc = dq >> 5, kk = dq & 31;
        *(v4h*)&sQA[nl >> 4][kc][laneA(nl & 15, kk)][slotA(kk)] = p;
    }

    float rk[4][4];   // prefetched K quads
    v4f   rv[3];      // prefetched V quads

    auto load_kv = [&](int m0) {
        #pragma unroll
        for (int i = 0; i < 4; ++i) {
            int e  = tid + i * 128;            // 512 quads (64d x 32 keys)
            int ml = e & 31, dq = (e >> 5) * 4;
            if (dq < 48) {
                const float* k4 = &kp[(size_t)dq * 1024 + m0 + ml];
                rk[i][0] = k4[0]; rk[i][1] = k4[1024];
                rk[i][2] = k4[2048]; rk[i][3] = k4[3072];
            } else {
                rk[i][0] = rk[i][1] = rk[i][2] = rk[i][3] = 0.f;
            }
        }
        #pragma unroll
        for (int i = 0; i < 3; ++i) {
            int e  = tid + i * 128;            // 384 quads (32 keys x 48 d)
            int mq = (e & 7) * 4, dd = e >> 3;
            rv[i] = *(const v4f*)&vp[(size_t)dd * 1024 + m0 + mq];
        }
    };
    auto store_kv = [&](int buf) {
        #pragma unroll
        for (int i = 0; i < 4; ++i) {
            int e  = tid + i * 128;
            int ml = e & 31, dq = (e >> 5) * 4;
            int kc = dq >> 5, kk = dq & 31;
            *(v4h*)&sKB[buf][kc][ml >> 4][laneB(kk, ml & 15)][slotB(kk)] =
                cvt4(rk[i][0], rk[i][1], rk[i][2], rk[i][3]);
        }
        #pragma unroll
        for (int i = 0; i < 3; ++i) {
            int e  = tid + i * 128;
            int mq = (e & 7) * 4, dd = e >> 3;
            *(v4h*)&sVB[buf][dd >> 4][laneB(mq, dd & 15)][slotB(mq)] =
                cvt4(rv[i][0], rv[i][1], rv[i][2], rv[i][3]);
        }
    };

    load_kv(0);
    store_kv(0);
    __syncthreads();                           // sQA also published here

    const v16h aq0 = load_frag(&sQA[w][0][lane][0]);
    const v16h aq1 = load_frag(&sQA[w][1][lane][0]);

    float mrow[8], lrow[8];
    #pragma unroll
    for (int v = 0; v < 8; ++v) { mrow[v] = -INFINITY; lrow[v] = 0.0f; }
    v8f O0 = {}, O1 = {}, O2 = {};

    for (int m0 = 0; m0 < 1024; m0 += 32) {
        const int cur = (m0 >> 5) & 1;
        const bool has_next = (m0 + 32) < 1024;
        if (has_next) load_kv(m0 + 32);        // VMEM overlaps S/softmax/O compute

        // S strip: 16 queries x 32 keys (K=64 padded, two WMMAs per 16x16 tile)
        v8f S0 = {}, S1 = {};
        S0 = WMMA_F32_F16(aq0, load_frag(&sKB[cur][0][0][lane][0]), S0);
        S0 = WMMA_F32_F16(aq1, load_frag(&sKB[cur][1][0][lane][0]), S0);
        S1 = WMMA_F32_F16(aq0, load_frag(&sKB[cur][0][1][lane][0]), S1);
        S1 = WMMA_F32_F16(aq1, load_frag(&sKB[cur][1][1][lane][0]), S1);

        // Online softmax; row r = v + 8g lives across the 16 lanes of half-group g
        #pragma unroll
        for (int v = 0; v < 8; ++v) {
            S0[v] *= scale; S1[v] *= scale;
            float mx = fmaxf(S0[v], S1[v]);
            #pragma unroll
            for (int off = 1; off < 16; off <<= 1) mx = fmaxf(mx, __shfl_xor(mx, off, 32));
            float mnew  = fmaxf(mrow[v], mx);
            float alpha = __expf(mrow[v] - mnew);
            mrow[v] = mnew;
            S0[v] = __expf(S0[v] - mnew);
            S1[v] = __expf(S1[v] - mnew);
            float rs = S0[v] + S1[v];
            #pragma unroll
            for (int off = 1; off < 16; off <<= 1) rs += __shfl_xor(rs, off, 32);
            lrow[v] = lrow[v] * alpha + rs;
            O0[v] *= alpha; O1[v] *= alpha; O2[v] *= alpha;
        }

        // C-layout -> A-layout scatter into per-wave fragment-ordered buffer
        #pragma unroll
        for (int v = 0; v < 8; ++v) {
            int r = v + 8 * g;
            sPA[w][16 * (ln >> 3) + r][ln & 7]       = (_Float16)S0[v];
            sPA[w][16 * (ln >> 3) + r][8 + (ln & 7)] = (_Float16)S1[v];
        }
        v16h ap = load_frag(&sPA[w][lane][0]);
        O0 = WMMA_F32_F16(ap, load_frag(&sVB[cur][0][lane][0]), O0);
        O1 = WMMA_F32_F16(ap, load_frag(&sVB[cur][1][lane][0]), O1);
        O2 = WMMA_F32_F16(ap, load_frag(&sVB[cur][2][lane][0]), O2);

        if (has_next) {
            store_kv(cur ^ 1);                 // cur^1 readers finished last barrier
            __syncthreads();                   // publish next buffer
        }
    }

    // out layout (B, 384, 1024), channel = head*48 + d
    #pragma unroll
    for (int v = 0; v < 8; ++v) {
        float inv = 1.0f / lrow[v];
        int n = qb + w * 16 + v + 8 * g;
        size_t base = ((size_t)b * 384 + head * 48) * 1024 + n;
        out[base + (size_t)(ln)      * 1024] = O0[v] * inv;
        out[base + (size_t)(16 + ln) * 1024] = O1[v] * inv;
        out[base + (size_t)(32 + ln) * 1024] = O2[v] * inv;
    }
}

// ---------------------------------------------------------------------------
// y = LayerNorm_channels(x + r) * w + b   (biased variance, eps = 1e-5)
// ---------------------------------------------------------------------------
__global__ __launch_bounds__(128)
void add_ln_kernel(const float* __restrict__ x, const float* __restrict__ r,
                   const float* __restrict__ w, const float* __restrict__ bb,
                   float* __restrict__ out, int C, int N, int total) {
    int pos = blockIdx.x * blockDim.x + threadIdx.x;
    if (pos >= total) return;
    int b = pos / N, n = pos % N;
    const float* xb = x + (size_t)b * C * N + n;
    const float* rb = r + (size_t)b * C * N + n;
    float s = 0.0f, s2 = 0.0f;
    for (int c = 0; c < C; ++c) {
        float v = xb[(size_t)c * N] + rb[(size_t)c * N];
        s += v; s2 += v * v;
    }
    float mean = s / C;
    float var  = s2 / C - mean * mean;
    float inv  = rsqrtf(var + 1e-5f);
    float* ob = out + (size_t)b * C * N + n;
    for (int c = 0; c < C; ++c) {
        float v = xb[(size_t)c * N] + rb[(size_t)c * N];
        ob[(size_t)c * N] = (v - mean) * inv * w[c] + bb[c];
    }
}

// ---------------------------------------------------------------------------
extern "C" void kernel_launch(void* const* d_in, const int* in_sizes, int n_in,
                              void* d_out, int out_size, void* d_ws, size_t ws_size,
                              hipStream_t stream) {
    const int B = 16, C = 384, N = 1024;
    const float* x      = (const float*)d_in[0];
    const float* w_qkv  = (const float*)d_in[1];
    const float* w_proj = (const float*)d_in[2];
    const float* ln1_w  = (const float*)d_in[3];
    const float* ln1_b  = (const float*)d_in[4];
    const float* w_fc1  = (const float*)d_in[5];
    const float* b_fc1  = (const float*)d_in[6];
    const float* w_fc2  = (const float*)d_in[7];
    const float* b_fc2  = (const float*)d_in[8];
    const float* ln2_w  = (const float*)d_in[9];
    const float* ln2_b  = (const float*)d_in[10];
    float* out = (float*)d_out;

    float* ws    = (float*)d_ws;
    float* qkv   = ws;                                  // 16*1152*1024
    float* attn  = qkv  + (size_t)B * 3 * C * N;        // 16*384*1024
    float* tmp   = attn + (size_t)B * C * N;            // proj out / fc2 out
    float* x1    = tmp  + (size_t)B * C * N;            // post-LN1 activations
    float* h1    = x1   + (size_t)B * C * N;            // 16*1536*1024

    const int ln_total = B * N;
    const float scale = 0.14433756729740643f;           // 48^-0.5

    // qkv = w_qkv @ x
    gemm_wmma_kernel<0><<<dim3(N / 128, (3 * C) / 64, B), 256, 0, stream>>>(
        w_qkv, x, nullptr, qkv, 3 * C, N, C);
    // attention
    attn_kernel<<<dim3(N / 64, 8, B), 128, 0, stream>>>(qkv, attn, scale);
    // proj
    gemm_wmma_kernel<0><<<dim3(N / 128, C / 64, B), 256, 0, stream>>>(
        w_proj, attn, nullptr, tmp, C, N, C);
    // x1 = LN1(x + proj)
    add_ln_kernel<<<(ln_total + 127) / 128, 128, 0, stream>>>(
        x, tmp, ln1_w, ln1_b, x1, C, N, ln_total);
    // h1 = gelu(fc1 @ x1 + b1)
    gemm_wmma_kernel<1><<<dim3(N / 128, (4 * C) / 64, B), 256, 0, stream>>>(
        w_fc1, x1, b_fc1, h1, 4 * C, N, C);
    // tmp = fc2 @ h1 + b2
    gemm_wmma_kernel<2><<<dim3(N / 128, C / 64, B), 256, 0, stream>>>(
        w_fc2, h1, b_fc2, tmp, C, N, 4 * C);
    // out = LN2(x1 + tmp)
    add_ln_kernel<<<(ln_total + 127) / 128, 128, 0, stream>>>(
        x1, tmp, ln2_w, ln2_b, out, C, N, ln_total);
}
